// IPOT_module_46832323396187
// MI455X (gfx1250) — matrix operand order repaired and checked by reading, not compiled, not run on hardware
//
#include <hip/hip_runtime.h>

// ---------------------------------------------------------------------------
// IPOT on MI455X (gfx1250, wave32).
//   T_k = exp(-k C) ∘ (r c^T) kept implicit -> no P / A matrices at all.
//   Iterations: two L2-resident sweeps of the 4MB C matrix per iter with
//   v_exp_f32 (TRANS, co-executes with VALU) regenerating A^k on the fly.
//   GEMMs: v_wmma_f32_16x16x32_f16. Each block owns a 16-row strip whose
//   f16 A-panel is staged into LDS by the Tensor Data Mover and reused by
//   8 waves x 8 tiles. Transposed outputs staged through LDS so all global
//   stores are contiguous 32B chunks.
// ---------------------------------------------------------------------------

#define N_TOK   1024
#define DIM     512
#define N_ITERS 50
#define NBLK    64            // 1024 / 16 tiles per dimension

typedef _Float16 v16h __attribute__((ext_vector_type(16)));
typedef _Float16 v8h  __attribute__((ext_vector_type(8)));
typedef float    v8f  __attribute__((ext_vector_type(8)));
typedef unsigned int u32x4 __attribute__((ext_vector_type(4)));
typedef int          i32x4 __attribute__((ext_vector_type(4)));
typedef int          i32x8 __attribute__((ext_vector_type(8)));

#if defined(__has_builtin)
# if __has_builtin(__builtin_amdgcn_tensor_load_to_lds) && __has_builtin(__builtin_amdgcn_s_wait_tensorcnt)
#  define HAVE_TDM 1
# endif
#endif
#ifndef HAVE_TDM
# define HAVE_TDM 0
#endif

union H16 { v16h v; v8h h[2]; };

// A-operand (16x32 f16, row-major source): lane holds row m = lane&15,
// contiguous K [b, b+16), b = k0 + (lane>=16 ? 16 : 0).
__device__ __forceinline__ v16h load_opA(const _Float16* rowbase, int k0, int lane) {
  H16 u;
  const int b = k0 + ((lane >> 4) << 4);
  u.h[0] = *(const v8h*)(rowbase + b);
  u.h[1] = *(const v8h*)(rowbase + b + 8);
  return u.v;
}

// B-operand (32x16 f16) for NT-style contraction: lane holds column n = lane&15
// (a row of the row-major source); contiguous K [k0 + (lane>=16 ? 16 : 0), +16).
__device__ __forceinline__ v16h load_opB(const _Float16* rowbase, int k0, int lane) {
  H16 u;
  const int b = k0 + ((lane >> 4) << 4);
  u.h[0] = *(const v8h*)(rowbase + b);
  u.h[1] = *(const v8h*)(rowbase + b + 8);
  return u.v;
}

#if HAVE_TDM
// TDM: stage a [rows x elems] f16 panel (row stride in elements, out of a
// tensor with tensor_rows rows) into LDS at byte offset lds_off.
// 2-D descriptor per CDNA5 ISA.
__device__ __forceinline__ void tdm_load_panel_f16(unsigned lds_off, const void* gptr,
                                                   unsigned elems, unsigned rows,
                                                   unsigned stride_elems, unsigned tensor_rows) {
  const unsigned long long ga = (unsigned long long)gptr;
  u32x4 g0;
  g0[0] = 1u;                                                  // count=1, user mode
  g0[1] = lds_off;                                             // lds_addr (bytes)
  g0[2] = (unsigned)ga;                                        // global_addr[31:0]
  g0[3] = (unsigned)((ga >> 32) & 0x01FFFFFFu) | (2u << 30);   // [56:32] | type=2
  i32x8 g1;
  g1[0] = (int)(1u << 16);                                     // data_size=1 (2B), mask=0
  g1[1] = (int)((elems & 0xFFFFu) << 16);                      // tensor_dim0[15:0]
  g1[2] = (int)(((elems >> 16) & 0xFFFFu) | ((tensor_rows & 0xFFFFu) << 16)); // dim0 hi | dim1 lo
  g1[3] = (int)(((tensor_rows >> 16) & 0xFFFFu) | ((elems & 0xFFFFu) << 16)); // dim1 hi | tile_dim0
  g1[4] = (int)(rows & 0xFFFFu);                               // tile_dim1 | tile_dim2=0
  g1[5] = (int)stride_elems;                                   // tensor_dim0_stride lo32
  g1[6] = 0;                                                   // stride hi | dim1_stride lo
  g1[7] = 0;
  i32x4 gz = {0, 0, 0, 0};
#if __clang_major__ >= 23
  i32x8 gz8 = {0, 0, 0, 0, 0, 0, 0, 0};
  __builtin_amdgcn_tensor_load_to_lds(g0, g1, gz, gz, gz8, 0);
#else
  __builtin_amdgcn_tensor_load_to_lds(g0, g1, gz, gz, 0);
#endif
}
#endif

// Stage a [16 x elems] f16 panel into LDS (TDM with cooperative fallback).
__device__ __forceinline__ void stage_panel(_Float16* sA, const _Float16* gsrc,
                                            unsigned elems) {
#if HAVE_TDM
  if (threadIdx.x < 32) {                                      // one wave issues DMA
    tdm_load_panel_f16(0u, gsrc, elems, 16u, elems, N_TOK);
    __builtin_amdgcn_s_wait_tensorcnt(0);
  }
  __syncthreads();
#else
  const v8h* src = (const v8h*)gsrc;
  v8h*       dst = (v8h*)sA;
  for (unsigned idx = threadIdx.x; idx < (16 * elems) / 8; idx += 256)
    dst[idx] = src[idx];
  __syncthreads();
#endif
}

// --------------------------- phase A: normalize ----------------------------
__global__ void k_normalize(const float* __restrict__ t, const float* __restrict__ v,
                            _Float16* __restrict__ Xh, _Float16* __restrict__ Yh) {
  __shared__ float red[256];
  const int row = blockIdx.x;
  const float* src = (row < N_TOK) ? (t + (size_t)row * DIM) : (v + (size_t)(row - N_TOK) * DIM);
  _Float16*    dst = (row < N_TOK) ? (Xh + (size_t)row * DIM) : (Yh + (size_t)(row - N_TOK) * DIM);
  float vals[2];
  float ss = 0.f;
#pragma unroll
  for (int k = 0; k < 2; ++k) {
    float x = src[threadIdx.x + 256 * k];
    vals[k] = x;
    ss += x * x;
  }
  red[threadIdx.x] = ss;
  __syncthreads();
  for (int s = 128; s > 0; s >>= 1) {
    if (threadIdx.x < s) red[threadIdx.x] += red[threadIdx.x + s];
    __syncthreads();
  }
  const float inv = rsqrtf(red[0]);
#pragma unroll
  for (int k = 0; k < 2; ++k)
    dst[threadIdx.x + 256 * k] = (_Float16)(vals[k] * inv);
}

// ------------- phase B: C = 1 - X Y^T  (f32) and C^T (f16) -----------------
// Block owns row-strip ti: X panel (16x512 f16, 16KB) staged to LDS via TDM;
// 8 waves x 8 tiles each. C^T staged through LDS -> contiguous 32B stores.
__global__ void k_gemm_C(const _Float16* __restrict__ Xh, const _Float16* __restrict__ Yh,
                         float* __restrict__ Cbuf, _Float16* __restrict__ CtH) {
  __shared__ _Float16 sA[16 * DIM];                            // 16KB A panel
  __shared__ _Float16 sCt[N_TOK * 16];                         // 32KB C^T staging
  const int lane = threadIdx.x & 31;
  const int wave = threadIdx.x >> 5;
  const int ti   = blockIdx.x;

  stage_panel(sA, Xh + (size_t)ti * 16 * DIM, DIM);

  const _Float16* arow = sA + (size_t)(lane & 15) * DIM;       // ds_load_b128 path
  const int rbase = ti * 16 + ((lane >> 4) << 3);
  const int rloc  = (lane >> 4) << 3;
#pragma unroll
  for (int q = 0; q < 2; ++q) {                                // 8 tiles per wave
    const int tjb = wave * 8 + q * 4;
    const _Float16* brow[4];
#pragma unroll
    for (int t = 0; t < 4; ++t)
      brow[t] = Yh + (size_t)((tjb + t) * 16 + (lane & 15)) * DIM;
    v8f acc[4] = {};
#pragma unroll 2
    for (int k0 = 0; k0 < DIM; k0 += 32) {
      const v16h a = load_opA(arow, k0, lane);
#pragma unroll
      for (int t = 0; t < 4; ++t) {
        const v16h b = load_opB(brow[t], k0, lane);
        acc[t] = __builtin_amdgcn_wmma_f32_16x16x32_f16(false, a, false, b, (short)0, acc[t], false, false);
      }
    }
#pragma unroll
    for (int t = 0; t < 4; ++t) {
      const int col = (tjb + t) * 16 + (lane & 15);
      v8h cvals;
#pragma unroll
      for (int vv = 0; vv < 8; ++vv) {
        const float Cv = 1.0f - acc[t][vv];
        Cbuf[(size_t)(rbase + vv) * N_TOK + col] = Cv;         // coalesced across lanes
        cvals[vv] = (_Float16)Cv;
      }
      *(v8h*)(sCt + (size_t)col * 16 + rloc) = cvals;          // LDS transpose staging
    }
  }
  __syncthreads();
  // contiguous 32B stores of C^T: column col -> CtH[col*N + ti*16 .. +16)
  const v8h* src = (const v8h*)sCt;
  for (int col = threadIdx.x; col < N_TOK; col += 256) {
    v8h* dst = (v8h*)(CtH + (size_t)col * N_TOK + ti * 16);
    dst[0] = src[col * 2 + 0];
    dst[1] = src[col * 2 + 1];
  }
}

// --------------------------- iteration kernels -----------------------------
__global__ void k_init_vec(float* __restrict__ c, float* __restrict__ w) {
  const int j = blockIdx.x * blockDim.x + threadIdx.x;
  c[j] = 1.0f;
  w[j] = 1.0f / (float)N_TOK;
}

// y = exp(negk*C) w ; r = 1/(n y).   (A^k regenerated on the fly: no P matrix)
__global__ void k_rowpass(const float* __restrict__ C, const float* __restrict__ w,
                          float* __restrict__ r, float negk) {
  __shared__ float red[256];
  const int i = blockIdx.x;
  const float* crow = C + (size_t)i * N_TOK;
  float acc = 0.f;
  for (int j = threadIdx.x; j < N_TOK; j += 256) {
    __builtin_prefetch(crow + j + 512, 0, 0);                  // global_prefetch
    acc += __expf(negk * crow[j]) * w[j];
  }
  red[threadIdx.x] = acc;
  __syncthreads();
  for (int s = 128; s > 0; s >>= 1) {
    if (threadIdx.x < s) red[threadIdx.x] += red[threadIdx.x + s];
    __syncthreads();
  }
  if (threadIdx.x == 0) r[i] = 1.0f / ((float)N_TOK * red[0]);
}

// z = exp(negk*C)^T r ; c_new = 1/(n z) ; w_new = c_new^2 / c_old.
__global__ void k_colpass(const float* __restrict__ C, const float* __restrict__ r,
                          float* __restrict__ c, float* __restrict__ w, float negk) {
  const int j = blockIdx.x * blockDim.x + threadIdx.x;         // one column / thread
  float acc = 0.f;
  for (int i = 0; i < N_TOK; ++i) {
    if ((i & 3) == 0) __builtin_prefetch(&C[(size_t)(i + 4) * N_TOK + j], 0, 0);
    acc += __expf(negk * C[(size_t)i * N_TOK + j]) * r[i];     // coalesced
  }
  const float cn   = 1.0f / ((float)N_TOK * acc);
  const float cold = c[j];
  c[j] = cn;
  w[j] = cn * (cn / cold);
}

// --------- phase D: T^T (scaled by n) in f16; T = exp(-50C) ∘ r c^T --------
// 16-row strip per block, transposed through LDS -> contiguous 32B stores.
__global__ void k_buildT(const float* __restrict__ C, const float* __restrict__ r,
                         const float* __restrict__ c, _Float16* __restrict__ TtH) {
  __shared__ _Float16 sT[N_TOK * 16];                          // 32KB staging
  const int i0 = blockIdx.x * 16;
  for (int e = threadIdx.x; e < 16 * N_TOK; e += 256) {
    const int ii = e >> 10;                                    // row in strip
    const int j  = e & (N_TOK - 1);                            // coalesced column
    const float val = __expf(-(float)N_ITERS * C[(size_t)(i0 + ii) * N_TOK + j])
                    * (r[i0 + ii] * (float)N_TOK) * c[j];      // n-scale for f16 range
    sT[(size_t)j * 16 + ii] = (_Float16)val;
  }
  __syncthreads();
  const v8h* src = (const v8h*)sT;
  for (int j = threadIdx.x; j < N_TOK; j += 256) {
    v8h* dst = (v8h*)(TtH + (size_t)j * N_TOK + i0);
    dst[0] = src[j * 2 + 0];
    dst[1] = src[j * 2 + 1];
  }
}

// ---------- phase E: out[b1,b2] = mean of 16x16 tile of C^T T --------------
// Block owns strip b1: 32KB C^T panel staged via TDM, reused by 8 waves x 8
// tiles; B streams from L2; block-mean fused into the epilogue.
__global__ void k_gemm_out(const _Float16* __restrict__ CtH, const _Float16* __restrict__ TtH,
                           float* __restrict__ out) {
  __shared__ _Float16 sA[16 * N_TOK];                          // 32KB panel
  const int lane = threadIdx.x & 31;
  const int wave = threadIdx.x >> 5;
  const int b1   = blockIdx.x;

  stage_panel(sA, CtH + (size_t)b1 * 16 * N_TOK, N_TOK);

  const _Float16* arow = sA + (size_t)(lane & 15) * N_TOK;     // ds_load_b128 path
#pragma unroll
  for (int q = 0; q < 2; ++q) {                                // 8 tiles per wave
    const int b2b = wave * 8 + q * 4;
    const _Float16* brow[4];
#pragma unroll
    for (int t = 0; t < 4; ++t)
      brow[t] = TtH + (size_t)((b2b + t) * 16 + (lane & 15)) * N_TOK;
    v8f acc[4] = {};
#pragma unroll 2
    for (int k0 = 0; k0 < N_TOK; k0 += 32) {
      const v16h a = load_opA(arow, k0, lane);
#pragma unroll
      for (int t = 0; t < 4; ++t) {
        const v16h b = load_opB(brow[t], k0, lane);
        acc[t] = __builtin_amdgcn_wmma_f32_16x16x32_f16(false, a, false, b, (short)0, acc[t], false, false);
      }
    }
#pragma unroll
    for (int t = 0; t < 4; ++t) {
      float s = 0.f;
#pragma unroll
      for (int vv = 0; vv < 8; ++vv) s += acc[t][vv];
      for (int m = 16; m >= 1; m >>= 1) s += __shfl_xor(s, m, 32);
      if (lane == 0)
        out[b1 * NBLK + (b2b + t)] = s * (1.0f / (256.0f * (float)N_TOK));
    }
  }
}

__global__ void k_trace(const float* __restrict__ out, float* __restrict__ loss) {
  __shared__ float red[64];
  const int d = threadIdx.x;
  red[d] = out[d * NBLK + d];
  __syncthreads();
  for (int s = 32; s > 0; s >>= 1) {
    if (d < s) red[d] += red[d + s];
    __syncthreads();
  }
  if (d == 0) loss[0] = red[0];
}

// ---------------------------------------------------------------------------
extern "C" void kernel_launch(void* const* d_in, const int* in_sizes, int n_in,
                              void* d_out, int out_size, void* d_ws, size_t ws_size,
                              hipStream_t stream) {
  (void)in_sizes; (void)n_in; (void)out_size; (void)ws_size;
  const float* t_prob = (const float*)d_in[0];
  const float* v_prob = (const float*)d_in[1];
  float* out = (float*)d_out;

  char* ws = (char*)d_ws;                                      // ~10.5 MB total
  size_t off = 0;
  auto carve = [&](size_t bytes) -> void* {
    void* p = ws + off;
    off = (off + bytes + 255) & ~(size_t)255;
    return p;
  };
  _Float16* Xh   = (_Float16*)carve((size_t)N_TOK * DIM   * 2);
  _Float16* Yh   = (_Float16*)carve((size_t)N_TOK * DIM   * 2);
  float*    Cbuf = (float*)   carve((size_t)N_TOK * N_TOK * 4);
  _Float16* CtH  = (_Float16*)carve((size_t)N_TOK * N_TOK * 2);
  _Float16* TtH  = (_Float16*)carve((size_t)N_TOK * N_TOK * 2);
  float*    wv   = (float*)   carve((size_t)N_TOK * 4);
  float*    cv   = (float*)   carve((size_t)N_TOK * 4);
  float*    rv   = (float*)   carve((size_t)N_TOK * 4);

  k_normalize<<<2 * N_TOK, 256, 0, stream>>>(t_prob, v_prob, Xh, Yh);
  k_gemm_C<<<NBLK, 256, 0, stream>>>(Xh, Yh, Cbuf, CtH);
  k_init_vec<<<N_TOK / 256, 256, 0, stream>>>(cv, wv);

  for (int it = 0; it < N_ITERS; ++it) {
    const float negk = -(float)(it + 1);                       // A^{k+1} = exp(-(k+1)C)
    k_rowpass<<<N_TOK, 256, 0, stream>>>(Cbuf, wv, rv, negk);
    k_colpass<<<N_TOK / 256, 256, 0, stream>>>(Cbuf, rv, cv, wv, negk);
  }

  k_buildT<<<NBLK, 256, 0, stream>>>(Cbuf, rv, cv, TtH);
  k_gemm_out<<<NBLK, 256, 0, stream>>>(CtH, TtH, out);
  k_trace<<<1, 64, 0, stream>>>(out, out + NBLK * NBLK);
}
